// CausalMHA_15599321219547
// MI455X (gfx1250) — matrix operand order repaired and checked by reading, not compiled, not run on hardware
//
#include <hip/hip_runtime.h>
#include <hip/hip_bf16.h>

typedef __attribute__((ext_vector_type(16))) __bf16 v16bf;
typedef __attribute__((ext_vector_type(8)))  __bf16 v8bf;
typedef __attribute__((ext_vector_type(8)))  float  v8f;

#define D_MODEL 1024
#define NHEAD   16
#define HDIM    64
#define TSEQ    2048
#define BATCH   2
#define MTOT    (BATCH*TSEQ)      // 4096
#define N_QKV   (3*D_MODEL)       // 3072
#define LOG2E   1.44269504088896f

// ---------- helpers ----------
static __device__ __forceinline__ v8bf ld8(const unsigned short* p) {
  return *reinterpret_cast<const v8bf*>(p);
}
static __device__ __forceinline__ v16bf mk16(v8bf lo, v8bf hi) {
  return __builtin_shufflevector(lo, hi, 0,1,2,3,4,5,6,7,8,9,10,11,12,13,14,15);
}
static __device__ __forceinline__ unsigned short f2bf(float f) {
  __bf16 h = (__bf16)f;
  unsigned short u;
  __builtin_memcpy(&u, &h, 2);
  return u;
}
static __device__ __forceinline__ v8f wmma_bf16(v16bf a, v16bf b, v8f c) {
  return __builtin_amdgcn_wmma_f32_16x16x32_bf16(false, a, false, b, (short)0, c, false, false);
}
// A-fragment: row-major source, row fixed per lane, chunks K = 8*kh+[0,8) and 16+8*kh+[0,8)
static __device__ __forceinline__ v16bf afrag(const unsigned short* row, int kb, int kh) {
  return mk16(ld8(row + kb + 8 * kh), ld8(row + kb + 16 + 8 * kh));
}
// B-fragment: transposed source, col fixed per lane, chunks K = 8*kh+[0,8) and 16+8*kh+[0,8)
static __device__ __forceinline__ v16bf bfrag(const unsigned short* row, int kb, int kh) {
  return mk16(ld8(row + kb + 8 * kh), ld8(row + kb + 16 + 8 * kh));
}

// 32x64 register-blocked GEMM core: 8 WMMAs per 12 b128 loads
static __device__ __forceinline__ void gemm32x64(
    const unsigned short* a0, const unsigned short* a1,
    const unsigned short* b0, const unsigned short* b1,
    const unsigned short* b2, const unsigned short* b3,
    int K, int kh, v8f acc[2][4]) {
#pragma unroll 2
  for (int kt = 0; kt < K / 32; ++kt) {
    const int kb = kt * 32;
    v16bf A0 = afrag(a0, kb, kh);
    v16bf A1 = afrag(a1, kb, kh);
    v16bf B0 = bfrag(b0, kb, kh);
    v16bf B1 = bfrag(b1, kb, kh);
    v16bf B2 = bfrag(b2, kb, kh);
    v16bf B3 = bfrag(b3, kb, kh);
    acc[0][0] = wmma_bf16(A0, B0, acc[0][0]);
    acc[0][1] = wmma_bf16(A0, B1, acc[0][1]);
    acc[0][2] = wmma_bf16(A0, B2, acc[0][2]);
    acc[0][3] = wmma_bf16(A0, B3, acc[0][3]);
    acc[1][0] = wmma_bf16(A1, B0, acc[1][0]);
    acc[1][1] = wmma_bf16(A1, B1, acc[1][1]);
    acc[1][2] = wmma_bf16(A1, B2, acc[1][2]);
    acc[1][3] = wmma_bf16(A1, B3, acc[1][3]);
  }
}

// ---------- conversion / packing kernels ----------
__global__ void conv_f32_to_bf16(const float* __restrict__ src,
                                 unsigned short* __restrict__ dst, int n) {
  int i = blockIdx.x * blockDim.x + threadIdx.x;
  if (i < n) dst[i] = f2bf(src[i]);
}

// src is [K,N] row-major fp32; dst is [N,K] row-major bf16
__global__ void conv_transpose_bf16(const float* __restrict__ src,
                                    unsigned short* __restrict__ dst,
                                    int K, int N) {
  int i = blockIdx.x * blockDim.x + threadIdx.x;
  if (i >= K * N) return;
  int k = i / N, n = i % N;
  dst[(size_t)n * K + k] = f2bf(src[i]);
}

// ---------- QKV GEMM: [4096,1024] x [1024,3072] + bias -> q/k/v(bf16) ----------
__global__ __launch_bounds__(256) void qkv_gemm(
    const unsigned short* __restrict__ Xb,   // [4096,1024] bf16
    const unsigned short* __restrict__ WT,   // [3072,1024] bf16 (transposed)
    const float* __restrict__ bias,          // [3072]
    unsigned short* __restrict__ Qh,         // [2,16,2048,64]
    unsigned short* __restrict__ Kh,         // [2,16,2048,64]
    unsigned short* __restrict__ VTh) {      // [2,16,64,2048]
  const int lane = threadIdx.x & 31;
  const int kh   = lane >> 4;
  const int mr   = lane & 15;
  const int wave = (blockIdx.x * blockDim.x + threadIdx.x) >> 5;
  const int tN   = wave % (N_QKV / 64);      // 48 col-blocks of 64
  const int tM   = wave / (N_QKV / 64);      // 128 row-blocks of 32

  const unsigned short* a0 = Xb + (size_t)(tM * 32 + mr) * D_MODEL;
  const unsigned short* a1 = Xb + (size_t)(tM * 32 + 16 + mr) * D_MODEL;
  const unsigned short* b0 = WT + (size_t)(tN * 64 + mr) * D_MODEL;
  const unsigned short* b1 = WT + (size_t)(tN * 64 + 16 + mr) * D_MODEL;
  const unsigned short* b2 = WT + (size_t)(tN * 64 + 32 + mr) * D_MODEL;
  const unsigned short* b3 = WT + (size_t)(tN * 64 + 48 + mr) * D_MODEL;

  v8f acc[2][4] = {};
  gemm32x64(a0, a1, b0, b1, b2, b3, D_MODEL, kh, acc);

#pragma unroll
  for (int j = 0; j < 4; ++j) {
    const int ncol  = tN * 64 + j * 16 + mr;
    const float bv  = bias[ncol];
    const int which = ncol >> 10;            // 0=q 1=k 2=v
    const int hcol  = ncol & 1023;
    const int h     = hcol >> 6;
    const int dd    = hcol & 63;
#pragma unroll
    for (int mi = 0; mi < 2; ++mi)
#pragma unroll
      for (int r = 0; r < 8; ++r) {
        const int row  = tM * 32 + mi * 16 + 8 * kh + r;
        const int bIdx = row >> 11;
        const int t    = row & 2047;
        const unsigned short val = f2bf(acc[mi][j][r] + bv);
        const size_t bh = (size_t)(bIdx * NHEAD + h);
        if (which == 0)      Qh [(bh * TSEQ + t) * HDIM + dd] = val;
        else if (which == 1) Kh [(bh * TSEQ + t) * HDIM + dd] = val;
        else                 VTh[(bh * HDIM + dd) * TSEQ + t] = val;
      }
  }
}

// ---------- causal flash attention: one wave per (b,h,16-query block) ----------
struct AttnState {
  v8f acc[4];
  float mrow[8], lrow[8];
};

template <bool MASK>
static __device__ __forceinline__ void attn_block(
    AttnState& st, const v16bf qf[2],
    const unsigned short* __restrict__ Kh,
    const unsigned short* __restrict__ VTh,
    unsigned short* __restrict__ pl,
    size_t bh, int jb, int qbase, int kh, int mr) {
  // ---- S = Q K^T for 32 keys (two 16-col tiles) ----
  v8f s0 = {}, s1 = {};
  const unsigned short* k0 = Kh + (bh * TSEQ + jb + mr) * HDIM;
  const unsigned short* k1 = Kh + (bh * TSEQ + jb + 16 + mr) * HDIM;
#pragma unroll
  for (int s = 0; s < 2; ++s) {
    v16bf B0 = bfrag(k0, s * 32, kh);
    v16bf B1 = bfrag(k1, s * 32, kh);
    s0 = wmma_bf16(qf[s], B0, s0);
    s1 = wmma_bf16(qf[s], B1, s1);
  }

  // ---- scale (+ causal mask) + online softmax (rows = 8*kh + r) ----
  float p0[8], p1[8];
#pragma unroll
  for (int r = 0; r < 8; ++r) {
    float v0 = s0[r] * 0.125f;               // 1/sqrt(64)
    float v1 = s1[r] * 0.125f;
    if (MASK) {
      const int qt = qbase + 8 * kh + r;
      if (jb + mr      > qt) v0 = -1e30f;
      if (jb + 16 + mr > qt) v1 = -1e30f;
    }
    float mx = fmaxf(v0, v1);
#pragma unroll
    for (int off = 1; off < 16; off <<= 1)
      mx = fmaxf(mx, __shfl_xor(mx, off, 32));
    const float mnew  = fmaxf(st.mrow[r], mx);
    const float alpha = exp2f((st.mrow[r] - mnew) * LOG2E);
    st.mrow[r] = mnew;
    const float e0 = exp2f((v0 - mnew) * LOG2E);
    const float e1 = exp2f((v1 - mnew) * LOG2E);
    float sum = e0 + e1;
#pragma unroll
    for (int off = 1; off < 16; off <<= 1)
      sum += __shfl_xor(sum, off, 32);
    st.lrow[r] = st.lrow[r] * alpha + sum;
#pragma unroll
    for (int nt = 0; nt < 4; ++nt) st.acc[nt][r] *= alpha;
    p0[r] = e0; p1[r] = e1;
  }

  // ---- stage P (16x32) in LDS: C/D layout -> A layout (per-wave DS is in-order) ----
#pragma unroll
  for (int r = 0; r < 8; ++r) {
    pl[(8 * kh + r) * 32 + mr]      = f2bf(p0[r]);
    pl[(8 * kh + r) * 32 + 16 + mr] = f2bf(p1[r]);
  }
  v16bf pa = mk16(ld8(pl + mr * 32 + 16 * kh), ld8(pl + mr * 32 + 16 * kh + 8));

  // ---- O += P V (V transposed: contiguous B-frags) ----
#pragma unroll
  for (int nt = 0; nt < 4; ++nt) {
    const unsigned short* vrow =
        VTh + (bh * HDIM + nt * 16 + mr) * TSEQ + jb + 16 * kh;
    v16bf bv = mk16(ld8(vrow), ld8(vrow + 8));
    st.acc[nt] = wmma_bf16(pa, bv, st.acc[nt]);
  }
}

__global__ __launch_bounds__(256) void flash_attn(
    const unsigned short* __restrict__ Qh,
    const unsigned short* __restrict__ Kh,
    const unsigned short* __restrict__ VTh,
    unsigned short* __restrict__ Ob) {       // [4096,1024] bf16
  __shared__ unsigned short ptile[8][16 * 32];   // per-wave P staging

  const int lane  = threadIdx.x & 31;
  const int kh    = lane >> 4;
  const int mr    = lane & 15;
  const int wIn   = threadIdx.x >> 5;
  const int wave  = blockIdx.x * 8 + wIn;
  const int qblk  = wave & 127;                  // T/16 = 128
  const int h     = (wave >> 7) & 15;
  const int bIdx  = wave >> 11;
  const int qbase = qblk * 16;
  const size_t bh = (size_t)(bIdx * NHEAD + h);

  // Q fragments (16 tokens x 64 dims -> two 16x32 A-frags)
  const unsigned short* qrow = Qh + (bh * TSEQ + qbase + mr) * HDIM;
  v16bf qf[2];
#pragma unroll
  for (int s = 0; s < 2; ++s) qf[s] = afrag(qrow, s * 32, kh);

  AttnState st;
#pragma unroll
  for (int nt = 0; nt < 4; ++nt) st.acc[nt] = (v8f){};
#pragma unroll
  for (int r = 0; r < 8; ++r) { st.mrow[r] = -1e30f; st.lrow[r] = 0.0f; }

  unsigned short* pl = ptile[wIn];
  const int jdiag = (qbase >> 5) << 5;     // the single block needing a mask
  for (int jb = 0; jb < jdiag; jb += 32)   // mask-free bulk
    attn_block<false>(st, qf, Kh, VTh, pl, bh, jb, qbase, kh, mr);
  attn_block<true>(st, qf, Kh, VTh, pl, bh, jdiag, qbase, kh, mr);

  // ---- normalize + store (reassemble heads into [B,T,D]) ----
#pragma unroll
  for (int nt = 0; nt < 4; ++nt)
#pragma unroll
    for (int r = 0; r < 8; ++r) {
      const int t = qbase + 8 * kh + r;
      Ob[((size_t)bIdx * TSEQ + t) * D_MODEL + h * HDIM + nt * 16 + mr] =
          f2bf(st.acc[nt][r] / st.lrow[r]);
    }
}

// ---------- output projection: [4096,1024] x [1024,1024] + bias -> fp32 ----------
__global__ __launch_bounds__(256) void proj_gemm(
    const unsigned short* __restrict__ Ab,   // [4096,1024] bf16
    const unsigned short* __restrict__ WT,   // [1024,1024] bf16 (transposed)
    const float* __restrict__ bias,          // [1024]
    float* __restrict__ out) {               // [4096,1024] fp32
  const int lane = threadIdx.x & 31;
  const int kh   = lane >> 4;
  const int mr   = lane & 15;
  const int wave = (blockIdx.x * blockDim.x + threadIdx.x) >> 5;
  const int tN   = wave % (D_MODEL / 64);    // 16 col-blocks of 64
  const int tM   = wave / (D_MODEL / 64);    // 128 row-blocks of 32

  const unsigned short* a0 = Ab + (size_t)(tM * 32 + mr) * D_MODEL;
  const unsigned short* a1 = Ab + (size_t)(tM * 32 + 16 + mr) * D_MODEL;
  const unsigned short* b0 = WT + (size_t)(tN * 64 + mr) * D_MODEL;
  const unsigned short* b1 = WT + (size_t)(tN * 64 + 16 + mr) * D_MODEL;
  const unsigned short* b2 = WT + (size_t)(tN * 64 + 32 + mr) * D_MODEL;
  const unsigned short* b3 = WT + (size_t)(tN * 64 + 48 + mr) * D_MODEL;

  v8f acc[2][4] = {};
  gemm32x64(a0, a1, b0, b1, b2, b3, D_MODEL, kh, acc);

#pragma unroll
  for (int j = 0; j < 4; ++j) {
    const int ncol = tN * 64 + j * 16 + mr;
    const float bv = bias[ncol];
#pragma unroll
    for (int mi = 0; mi < 2; ++mi)
#pragma unroll
      for (int r = 0; r < 8; ++r) {
        const int row = tM * 32 + mi * 16 + 8 * kh + r;
        out[(size_t)row * D_MODEL + ncol] = acc[mi][j][r] + bv;  // coalesced over 16 lanes
      }
  }
}

// ---------- host launcher ----------
extern "C" void kernel_launch(void* const* d_in, const int* in_sizes, int n_in,
                              void* d_out, int out_size, void* d_ws, size_t ws_size,
                              hipStream_t stream) {
  const float* x      = (const float*)d_in[0];
  const float* W_kqv  = (const float*)d_in[1];
  const float* b_kqv  = (const float*)d_in[2];
  const float* W_proj = (const float*)d_in[3];
  const float* b_proj = (const float*)d_in[4];
  float* out = (float*)d_out;

  char* ws = (char*)d_ws;
  size_t off = 0;
  unsigned short* Xb  = (unsigned short*)(ws + off); off += (size_t)MTOT * D_MODEL * 2;    // 8 MB
  unsigned short* WkT = (unsigned short*)(ws + off); off += (size_t)N_QKV * D_MODEL * 2;   // 6 MB
  unsigned short* WpT = (unsigned short*)(ws + off); off += (size_t)D_MODEL * D_MODEL * 2; // 2 MB
  unsigned short* Qh  = (unsigned short*)(ws + off); off += (size_t)MTOT * D_MODEL * 2;    // 8 MB
  unsigned short* Kh  = (unsigned short*)(ws + off); off += (size_t)MTOT * D_MODEL * 2;    // 8 MB
  unsigned short* VTh = (unsigned short*)(ws + off); off += (size_t)MTOT * D_MODEL * 2;    // 8 MB
  unsigned short* Ob  = (unsigned short*)(ws + off); off += (size_t)MTOT * D_MODEL * 2;    // 8 MB

  conv_f32_to_bf16<<<(MTOT * D_MODEL) / 256, 256, 0, stream>>>(x, Xb, MTOT * D_MODEL);
  conv_transpose_bf16<<<(D_MODEL * N_QKV) / 256, 256, 0, stream>>>(W_kqv, WkT, D_MODEL, N_QKV);
  conv_transpose_bf16<<<(D_MODEL * D_MODEL) / 256, 256, 0, stream>>>(W_proj, WpT, D_MODEL, D_MODEL);

  // (4096/32)*(3072/64) = 6144 wave-tiles / 8 waves per block
  qkv_gemm<<<6144 / 8, 256, 0, stream>>>(Xb, WkT, b_kqv, Qh, Kh, VTh);

  // B*H*(T/16) = 4096 waves / 8 per block
  flash_attn<<<4096 / 8, 256, 0, stream>>>(Qh, Kh, VTh, Ob);

  // (4096/32)*(1024/64) = 2048 wave-tiles / 8 waves per block
  proj_gemm<<<2048 / 8, 256, 0, stream>>>(Ob, WpT, b_proj, out);
}
